// MeshConv_11802570130356
// MI455X (gfx1250) — compile-verified, matrix operand order checked by reading, not run on hardware
//
#include <hip/hip_runtime.h>
#include <stdint.h>

#define NVV 40962      // vertices
#define NFF 81920      // faces
#define BC  256        // BS * CIN
#define CINC 32
#define COUTC 64
#define KDIM 128       // CIN * 4
#define APAD 136       // 128 + 8 bf16 pad -> 272B row stride, conflict-free frags
#define BPAD 136

typedef __attribute__((ext_vector_type(16))) __bf16 v16bf;
typedef __attribute__((ext_vector_type(8)))  float  v8f;
typedef __attribute__((ext_vector_type(2)))  float  f2v;

union FragU { uint4 q[2]; v16bf v; };

static __device__ __forceinline__ unsigned short f2bf(float x) {
    unsigned int u = __float_as_uint(x);
    u += 0x7FFFu + ((u >> 16) & 1u);      // round-to-nearest-even
    return (unsigned short)(u >> 16);
}
static __device__ __forceinline__ float bf2f(unsigned short h) {
    return __uint_as_float(((unsigned int)h) << 16);
}

// ---------------------------------------------------------------------------
// Kernel 1: transpose input (256, NV) -> Xt (NV, 256) so gathers are 1KB
// contiguous rows (coalesced, L2-resident: 42MB < 192MB L2).
// ---------------------------------------------------------------------------
__global__ void __launch_bounds__(256) transpose_kernel(
    const float* __restrict__ in, float* __restrict__ out, int nv)
{
    __shared__ float tile[32][33];
    const int v0  = blockIdx.x * 32;
    const int bc0 = blockIdx.y * 32;
    const int tx  = threadIdx.x & 31;
    const int ty  = threadIdx.x >> 5;   // 0..7
    #pragma unroll
    for (int i = ty; i < 32; i += 8) {
        const int v = v0 + tx;
        tile[i][tx] = (v < nv) ? in[(size_t)(bc0 + i) * nv + v] : 0.f;
    }
    __syncthreads();
    #pragma unroll
    for (int i = ty; i < 32; i += 8) {
        const int v = v0 + i;
        if (v < nv) out[(size_t)v * BC + (bc0 + tx)] = tile[tx][i];
    }
}

// ---------------------------------------------------------------------------
// Kernel 2: fused G-SpMM + EW/NS contraction.
// Row r = d*NF + f of G has its 3 nnz at [3r, 3r+3). For each face f:
//   ewf[f][bc] = sum_d EW[f,d] * sum_j G_vals * Xt[G_cols][bc]   (bf16 out)
// Thread t = bc lane -> every gather reads 256 consecutive floats.
// ---------------------------------------------------------------------------
__global__ void __launch_bounds__(256) face_grad_kernel(
    const float* __restrict__ Xt,
    const int*   __restrict__ G_cols, const float* __restrict__ G_vals,
    const float* __restrict__ EW,     const float* __restrict__ NS,
    unsigned short* __restrict__ EWf, unsigned short* __restrict__ NSf)
{
    const int t  = threadIdx.x;           // bc
    const int f0 = blockIdx.x * 32;
    for (int fi = 0; fi < 32; ++fi) {
        const int f = f0 + fi;
        if (fi + 1 < 32) {
            __builtin_prefetch(&G_cols[3 * (size_t)(f + 1)], 0, 0);
            __builtin_prefetch(&EW[(size_t)(f + 1) * 3], 0, 0);
        }
        float acc_ew = 0.f, acc_ns = 0.f;
        #pragma unroll
        for (int d = 0; d < 3; ++d) {
            const size_t base = 3u * ((size_t)d * NFF + (size_t)f);
            float s = 0.f;
            #pragma unroll
            for (int j = 0; j < 3; ++j) {
                const int   col = G_cols[base + j];
                const float val = G_vals[base + j];
                s += val * Xt[(size_t)col * BC + t];
            }
            acc_ew += EW[(size_t)f * 3 + d] * s;
            acc_ns += NS[(size_t)f * 3 + d] * s;
        }
        EWf[(size_t)f * BC + t] = f2bf(acc_ew);
        NSf[(size_t)f * BC + t] = f2bf(acc_ns);
    }
}

// ---------------------------------------------------------------------------
// Kernel 3: fused feat construction (identity / laplacian / F-SpMM ew,ns)
// into an LDS bf16 A-tile (128 rows = 8 batches x 16 verts, K=128 = c*4+k),
// then GEMM vs coeffs (K=128, N=64) with v_wmma_f32_16x16x32_bf16.
// out[b,o,v] = sum_{c,k} feat[b,c,v,k] * coeffs[c*4+k, o]
// Wave w (of 8) handles batch b=w: 4 N-tiles x 4 K-steps = 16 WMMAs.
// A-fragments hoisted (invariant across N-tiles); D stored as 4x b64
// (8 consecutive v per lane) under a uniform full-tile guard.
// ---------------------------------------------------------------------------
__global__ void __launch_bounds__(256) feat_gemm_kernel(
    const float* __restrict__ Xt,
    const int*   __restrict__ L_cols, const float* __restrict__ L_vals,
    const int*   __restrict__ F_cols, const float* __restrict__ F_vals,
    const unsigned short* __restrict__ EWf,
    const unsigned short* __restrict__ NSf,
    const float* __restrict__ coeffs,   // (128, 64) f32
    float* __restrict__ out)            // (8, 64, NV) f32
{
    __shared__ __align__(16) unsigned short A_lds[128 * APAD];
    __shared__ __align__(16) unsigned short B_lds[COUTC * BPAD];

    const int t  = threadIdx.x;
    const int v0 = blockIdx.x * 16;

    // --- load coeffs -> bf16, column(n)-major so B frags are 16B LDS reads
    for (int idx = t; idx < KDIM * COUTC; idx += 256) {
        const int n  = idx >> 7;       // 0..63
        const int kk = idx & 127;      // 0..127
        B_lds[n * BPAD + kk] = f2bf(coeffs[(size_t)kk * COUTC + n]);
    }

    // --- build A tile: thread t owns bc = t for all 16 vertices
    const int b = t >> 5;              // batch 0..7
    const int c = t & 31;              // channel 0..31
    for (int vi = 0; vi < 16; ++vi) {
        int v = v0 + vi;
        if (v >= NVV) v = NVV - 1;     // clamp tail (stores masked later)
        if (vi + 1 < 16) {
            __builtin_prefetch(&L_cols[(size_t)(v + 1) * 7], 0, 0);
            __builtin_prefetch(&F_cols[(size_t)(v + 1) * 6], 0, 0);
        }
        // k=0: identity
        const float fid = Xt[(size_t)v * BC + t];
        // k=1: laplacian (7 nnz/row, structural)
        float lap = 0.f;
        {
            const size_t base = (size_t)v * 7;
            #pragma unroll
            for (int j = 0; j < 7; ++j)
                lap += L_vals[base + j] * Xt[(size_t)L_cols[base + j] * BC + t];
        }
        // k=2,3: F-SpMM of face ew/ns (6 nnz/row, shared indices)
        float gev = 0.f, gns = 0.f;
        {
            const size_t base = (size_t)v * 6;
            #pragma unroll
            for (int j = 0; j < 6; ++j) {
                const int   col = F_cols[base + j];
                const float val = F_vals[base + j];
                gev += val * bf2f(EWf[(size_t)col * BC + t]);
                gns += val * bf2f(NSf[(size_t)col * BC + t]);
            }
        }
        const int row = b * 16 + vi;
        const unsigned long long pack =
              (unsigned long long)f2bf(fid)
            | ((unsigned long long)f2bf(lap) << 16)
            | ((unsigned long long)f2bf(gev) << 32)
            | ((unsigned long long)f2bf(gns) << 48);
        *(unsigned long long*)(&A_lds[row * APAD + c * 4]) = pack;   // ds_store_b64
    }

    __syncthreads();

    // --- WMMA phase: wave = batch
    const int wave = t >> 5;
    const int lane = t & 31;
    const int half = lane >> 4;                    // lane half selects K sub-chunk
    const unsigned short* arow = &A_lds[(wave * 16 + (lane & 15)) * APAD];

    // A fragments: invariant across N-tiles -> load once (4 x 8 VGPRs)
    FragU afr[4];
    #pragma unroll
    for (int ks = 0; ks < 4; ++ks) {
        const int K0 = ks * 32;
        afr[ks].q[0] = *(const uint4*)(arow + K0 +      half * 8);
        afr[ks].q[1] = *(const uint4*)(arow + K0 + 16 + half * 8);
    }

    const bool full = (v0 + 16 <= NVV);            // uniform: fast path all blocks but last

    #pragma unroll
    for (int nt = 0; nt < 4; ++nt) {
        v8f acc = {};
        const int n = nt * 16 + (lane & 15);
        const unsigned short* bcol = &B_lds[n * BPAD];
        #pragma unroll
        for (int ks = 0; ks < 4; ++ks) {
            const int K0 = ks * 32;
            FragU bfr;
            bfr.q[0] = *(const uint4*)(bcol + K0 +      half * 8);
            bfr.q[1] = *(const uint4*)(bcol + K0 + 16 + half * 8);
            acc = __builtin_amdgcn_wmma_f32_16x16x32_bf16(
                false, afr[ks].v, false, bfr.v, (short)0, acc, false, false);
        }
        // D layout: VGPR g, lanes 0-15 -> M=g, lanes 16-31 -> M=g+8; N = lane%16
        // => per lane, acc[0..7] are 8 consecutive v at row (wave,o=n).
        float* op = out + ((size_t)wave * COUTC + n) * NVV + (v0 + half * 8);
        if (full) {
            #pragma unroll
            for (int g2 = 0; g2 < 4; ++g2) {       // 4x global_store_b64, nontemporal
                f2v s2;
                s2.x = acc[2 * g2];
                s2.y = acc[2 * g2 + 1];
                __builtin_nontemporal_store(s2, (f2v*)(op + 2 * g2));
            }
        } else {
            #pragma unroll
            for (int g = 0; g < 8; ++g) {
                const int v = v0 + half * 8 + g;
                if (v < NVV) op[g] = acc[g];
            }
        }
    }
}

extern "C" void kernel_launch(void* const* d_in, const int* in_sizes, int n_in,
                              void* d_out, int out_size, void* d_ws, size_t ws_size,
                              hipStream_t stream)
{
    const float* input  = (const float*)d_in[0];
    const int*   G_cols = (const int*)  d_in[2];
    const float* G_vals = (const float*)d_in[3];
    const int*   L_cols = (const int*)  d_in[5];
    const float* L_vals = (const float*)d_in[6];
    const int*   F_cols = (const int*)  d_in[8];
    const float* F_vals = (const float*)d_in[9];
    const float* EW     = (const float*)d_in[10];
    const float* NS     = (const float*)d_in[11];
    const float* coeffs = (const float*)d_in[12];
    float* out = (float*)d_out;

    char* ws = (char*)d_ws;
    size_t off = 0;
    float* Xt = (float*)(ws + off);
    off += (size_t)NVV * BC * sizeof(float);
    off  = (off + 255) & ~(size_t)255;
    unsigned short* EWf = (unsigned short*)(ws + off);
    off += (size_t)NFF * BC * sizeof(unsigned short);
    off  = (off + 255) & ~(size_t)255;
    unsigned short* NSf = (unsigned short*)(ws + off);

    dim3 tgrid((NVV + 31) / 32, BC / 32);
    transpose_kernel<<<tgrid, 256, 0, stream>>>(input, Xt, NVV);

    face_grad_kernel<<<NFF / 32, 256, 0, stream>>>(
        Xt, G_cols, G_vals, EW, NS, EWf, NSf);

    feat_gemm_kernel<<<(NVV + 15) / 16, 256, 0, stream>>>(
        Xt, L_cols, L_vals, F_cols, F_vals, EWf, NSf, coeffs, out);
}
